// StyleAttention_56573309223444
// MI455X (gfx1250) — matrix operand order repaired and checked by reading, compiled, not run on hardware
//
#include <hip/hip_runtime.h>
#include <hip/hip_bf16.h>

// ---------------------------------------------------------------------------
// StyleAttention for MI455X (gfx1250): bf16 WMMA projections + flash attention.
// Shapes: B=2, C=512, N=48*48=2304, H=8 heads, d=64.
// Roofline: ~31 GFLOP total, ~40 MB working set (fits L2) -> matrix-core bound;
// flash streaming avoids the 340 MB N^2 attention matrix entirely.
// ---------------------------------------------------------------------------

typedef __attribute__((ext_vector_type(16))) __bf16 v16bf;
typedef __attribute__((ext_vector_type(8)))  __bf16 v8bf;
typedef __attribute__((ext_vector_type(8)))  float  v8f;

#define WMMA_BF16(A, B, C) \
  __builtin_amdgcn_wmma_f32_16x16x32_bf16(false, (A), false, (B), (short)0, (C), false, false)

#define B_N   2
#define C_N   512
#define H_N   8
#define D_N   64
#define N_N   2304   // 48*48

// ---------------------------------------------------------------------------
// Projection GEMM: Y = (W @ X + bias) * scale, W:(512,512) f32, X:(B,512,N).
// Each wave computes a 64(cout) x 16(n) tile: 4 A-frags of W share 1 B-frag
// of X per K=32 step -> 4 v_wmma_f32_16x16x32_bf16 per step, 16 steps.
// MODE 0: bf16 out, layout [b][h][n][d] (for Q/K; attention A/B frags load
//         contiguously from this). MODE 1: bf16 out, [b][c][n] (for V).
// MODE 2: f32 out, [b][c][n] (final projection -> d_out).
// ---------------------------------------------------------------------------
template <typename TIN, int MODE>
__global__ __launch_bounds__(256) void proj_kernel(
    const float* __restrict__ W, const float* __restrict__ bias,
    const TIN* __restrict__ X, void* __restrict__ Y, float scale)
{
  const int lane = threadIdx.x & 31;
  const int half = lane >> 4;     // 0: lanes 0-15, 1: lanes 16-31
  const int lq   = lane & 15;

  const int wid = blockIdx.x * 8 + (threadIdx.x >> 5);   // 288 blocks * 8 waves
  const int b   = wid / (8 * 144);
  const int rem = wid % (8 * 144);
  const int m0  = (rem / 144) * 64;    // cout tile base
  const int n0  = (rem % 144) * 16;    // token tile base

  v8f acc[4];
#pragma unroll
  for (int t = 0; t < 4; ++t)
#pragma unroll
    for (int i = 0; i < 8; ++i) acc[t][i] = 0.0f;

  const size_t xbase = (size_t)b * C_N * N_N;

  for (int kb = 0; kb < C_N; kb += 32) {
    // ---- B fragment from X: element(K=cin, N=n); lane holds col n0+lq,
    //      e[0..7] <-> K = half*8 + idx, e[8..15] <-> K = 16 + half*8 + idx-8.
    v16bf bx;
#pragma unroll
    for (int idx = 0; idx < 16; ++idx) {
      const int cin = kb + (idx < 8 ? half * 8 + idx : 16 + half * 8 + (idx - 8));
      const float xv = (float)X[xbase + (size_t)cin * N_N + n0 + lq];
      bx[idx] = (__bf16)xv;
    }
#pragma unroll
    for (int mt = 0; mt < 4; ++mt) {
      // ---- A fragment from W: row M = lq, e[0..7] = K half*8+0..7,
      //      e[8..15] = K 16+half*8+0..7 (two contiguous 8-float runs).
      const float* wrow = W + (size_t)(m0 + mt * 16 + lq) * C_N + kb;
      const float4 lo0 = *(const float4*)(wrow + half * 8);
      const float4 lo1 = *(const float4*)(wrow + half * 8 + 4);
      const float4 hi0 = *(const float4*)(wrow + 16 + half * 8);
      const float4 hi1 = *(const float4*)(wrow + 16 + half * 8 + 4);
      v16bf a;
      a[0]  = (__bf16)lo0.x; a[1]  = (__bf16)lo0.y; a[2]  = (__bf16)lo0.z; a[3]  = (__bf16)lo0.w;
      a[4]  = (__bf16)lo1.x; a[5]  = (__bf16)lo1.y; a[6]  = (__bf16)lo1.z; a[7]  = (__bf16)lo1.w;
      a[8]  = (__bf16)hi0.x; a[9]  = (__bf16)hi0.y; a[10] = (__bf16)hi0.z; a[11] = (__bf16)hi0.w;
      a[12] = (__bf16)hi1.x; a[13] = (__bf16)hi1.y; a[14] = (__bf16)hi1.z; a[15] = (__bf16)hi1.w;
      acc[mt] = WMMA_BF16(a, bx, acc[mt]);
    }
  }

  // ---- bias + scale + store. C-layout: VGPR r <-> row M = r + 8*half,
  //      lane col N = lq.
#pragma unroll
  for (int mt = 0; mt < 4; ++mt) {
#pragma unroll
    for (int r = 0; r < 8; ++r) {
      const int cout = m0 + mt * 16 + r + 8 * half;
      const int n    = n0 + lq;
      const float val = (acc[mt][r] + bias[cout]) * scale;
      if (MODE == 0) {
        const int h = cout >> 6, dd = cout & 63;
        ((__bf16*)Y)[(((size_t)b * H_N + h) * N_N + n) * D_N + dd] = (__bf16)val;
      } else if (MODE == 1) {
        ((__bf16*)Y)[((size_t)b * C_N + cout) * N_N + n] = (__bf16)val;
      } else {
        ((float*)Y)[((size_t)b * C_N + cout) * N_N + n] = val;
      }
    }
  }
}

// ---------------------------------------------------------------------------
// Flash attention: 8 waves/block, 16 queries/wave, 64 keys per step.
// Q pre-scaled by 0.125*log2(e) so softmax runs in exp2 domain (native
// v_exp_f32, co-executes with XDL wmma). Per step: 8 wmma for S (4 key tiles
// x K-dim 64), ONE online-softmax update per row per 64 keys (16-lane
// shfl_xor reductions + accumulator rescale amortized 2x vs a 32-key step),
// P round-trips per-wave LDS (C-layout -> A-layout), 8 wmma for P*V.
// ---------------------------------------------------------------------------
__global__ __launch_bounds__(256) void attn_kernel(
    const __bf16* __restrict__ Q,   // [b][h][n][d], pre-scaled
    const __bf16* __restrict__ K,   // [b][h][n][d]
    const __bf16* __restrict__ V,   // [b][c][n]
    __bf16* __restrict__ O)         // [b][c][n]
{
  __shared__ __bf16 pbuf[8][16 * 64];   // per-wave P staging (2 KB each)

  const int lane = threadIdx.x & 31;
  const int half = lane >> 4;
  const int lq   = lane & 15;
  const int w    = threadIdx.x >> 5;
  const int bh   = blockIdx.y;                   // b*8 + h
  const int b    = bh >> 3, h = bh & 7;
  const int q0   = blockIdx.x * 128 + w * 16;

  const __bf16* qb = Q + (size_t)bh * N_N * D_N;
  const __bf16* kp = K + (size_t)bh * N_N * D_N;
  const __bf16* vb = V + ((size_t)b * C_N + h * D_N) * N_N;

  // ---- preload Q A-fragments (d split 0..31 / 32..63), contiguous b128 loads
  v16bf aq[2];
  {
    const __bf16* qrow = qb + (size_t)(q0 + lq) * D_N;
#pragma unroll
    for (int kd = 0; kd < 2; ++kd) {
      const v8bf lo = *(const v8bf*)(qrow + kd * 32 + half * 16);
      const v8bf hi = *(const v8bf*)(qrow + kd * 32 + half * 16 + 8);
#pragma unroll
      for (int i = 0; i < 8; ++i) { aq[kd][i] = lo[i]; aq[kd][8 + i] = hi[i]; }
    }
  }

  v8f o[4];
#pragma unroll
  for (int t = 0; t < 4; ++t)
#pragma unroll
    for (int i = 0; i < 8; ++i) o[t][i] = 0.0f;
  float mrun[8], lrun[8];
#pragma unroll
  for (int r = 0; r < 8; ++r) { mrun[r] = -1e30f; lrun[r] = 0.0f; }

  for (int nk0 = 0; nk0 < N_N; nk0 += 64) {
    // ---- prefetch next step's K rows (wave-uniform branch; EXEC stays ~0xffffffff)
    if (nk0 + 64 < N_N) {
      __builtin_prefetch(kp + (size_t)(nk0 + 64 + lq) * D_N, 0, 1);
      __builtin_prefetch(vb + (size_t)lq * N_N + nk0 + 64, 0, 1);
    }

    // ---- S tiles: 4 tiles of 16 keys, K-dim = d = 64 (2 wmma each)
    v8f s[4];
#pragma unroll
    for (int t = 0; t < 4; ++t)
#pragma unroll
      for (int i = 0; i < 8; ++i) s[t][i] = 0.0f;
#pragma unroll
    for (int t = 0; t < 4; ++t) {
      const __bf16* krow = kp + (size_t)(nk0 + 16 * t + lq) * D_N;
#pragma unroll
      for (int kd = 0; kd < 2; ++kd) {
        const v16bf bk = *(const v16bf*)(krow + kd * 32 + half * 16);
        s[t] = WMMA_BF16(aq[kd], bk, s[t]);
      }
    }

    // ---- online softmax (base-2 domain), ONE update per row per 64 keys.
    //      Row M = r + 8*half lives in the 16 lanes of this half.
#pragma unroll
    for (int r = 0; r < 8; ++r) {
      float v = fmaxf(fmaxf(s[0][r], s[1][r]), fmaxf(s[2][r], s[3][r]));
#pragma unroll
      for (int m = 8; m >= 1; m >>= 1) v = fmaxf(v, __shfl_xor(v, m, 16));
      const float nm    = fmaxf(mrun[r], v);
      const float alpha = exp2f(mrun[r] - nm);
      float p[4];
#pragma unroll
      for (int t = 0; t < 4; ++t) p[t] = exp2f(s[t][r] - nm);
      float rs = (p[0] + p[1]) + (p[2] + p[3]);
#pragma unroll
      for (int m = 8; m >= 1; m >>= 1) rs += __shfl_xor(rs, m, 16);
      lrun[r] = lrun[r] * alpha + rs;
      mrun[r] = nm;
#pragma unroll
      for (int dt = 0; dt < 4; ++dt) o[dt][r] *= alpha;
      const int row = r + 8 * half;
#pragma unroll
      for (int t = 0; t < 4; ++t)
        pbuf[w][row * 64 + 16 * t + lq] = (__bf16)p[t];
    }

    // ---- P A-fragments from LDS: keys [0,32) and [32,64) of this step
    //      (same-wave LDS is in-order; compiler inserts s_wait_dscnt)
    v16bf ap[2];
#pragma unroll
    for (int kk = 0; kk < 2; ++kk) {
      const __bf16* prow = &pbuf[w][lq * 64 + kk * 32];
      const v8bf lo = *(const v8bf*)(prow + half * 16);
      const v8bf hi = *(const v8bf*)(prow + half * 16 + 8);
#pragma unroll
      for (int i = 0; i < 8; ++i) { ap[kk][i] = lo[i]; ap[kk][8 + i] = hi[i]; }
    }

    // ---- O += P V : B-frag element(K=key, N=dd) contiguous from [c][n]
#pragma unroll
    for (int dt = 0; dt < 4; ++dt) {
      const __bf16* vrow = vb + (size_t)(dt * 16 + lq) * N_N + nk0;
#pragma unroll
      for (int kk = 0; kk < 2; ++kk) {
        const v16bf bv = *(const v16bf*)(vrow + kk * 32 + half * 16);
        o[dt] = WMMA_BF16(ap[kk], bv, o[dt]);
      }
    }
  }

  // ---- normalize rows and store O as [b][c][n] (c = h*64 + dd, n = query)
#pragma unroll
  for (int dt = 0; dt < 4; ++dt) {
#pragma unroll
    for (int r = 0; r < 8; ++r) {
      const float inv = 1.0f / lrun[r];
      const size_t c = (size_t)b * C_N + h * D_N + dt * 16 + lq;
      O[c * N_N + q0 + r + 8 * half] = (__bf16)(o[dt][r] * inv);
    }
  }
}

// ---------------------------------------------------------------------------
// Launch: q/k/v projections -> flash attention -> output projection.
// Workspace: 4 bf16 buffers of B*C*N = 2,359,296 elements (18.9 MB total).
// ---------------------------------------------------------------------------
extern "C" void kernel_launch(void* const* d_in, const int* in_sizes, int n_in,
                              void* d_out, int out_size, void* d_ws, size_t ws_size,
                              hipStream_t stream) {
  const float* content = (const float*)d_in[0];
  const float* style   = (const float*)d_in[1];
  const float* Wq = (const float*)d_in[2];
  const float* bq = (const float*)d_in[3];
  const float* Wk = (const float*)d_in[4];
  const float* bk = (const float*)d_in[5];
  const float* Wv = (const float*)d_in[6];
  const float* bv = (const float*)d_in[7];
  const float* Wp = (const float*)d_in[8];
  const float* bp = (const float*)d_in[9];

  const size_t SZ = (size_t)B_N * C_N * N_N;   // elements per bf16 buffer
  __bf16* q_ws = (__bf16*)d_ws;
  __bf16* k_ws = q_ws + SZ;
  __bf16* v_ws = k_ws + SZ;
  __bf16* o_ws = v_ws + SZ;

  // softmax scale folded into Q, in base-2 domain: 1/sqrt(64) * log2(e)
  const float qscale = 0.125f * 1.44269504088896340736f;

  const dim3 pgrid(288), pblk(256);
  proj_kernel<float, 0><<<pgrid, pblk, 0, stream>>>(Wq, bq, content, (void*)q_ws, qscale);
  proj_kernel<float, 0><<<pgrid, pblk, 0, stream>>>(Wk, bk, style,   (void*)k_ws, 1.0f);
  proj_kernel<float, 1><<<pgrid, pblk, 0, stream>>>(Wv, bv, style,   (void*)v_ws, 1.0f);

  attn_kernel<<<dim3(18, 16), 256, 0, stream>>>(q_ws, k_ws, v_ws, o_ws);

  proj_kernel<__bf16, 2><<<pgrid, pblk, 0, stream>>>(Wp, bp, o_ws, d_out, 1.0f);
}
